// GNNEncoder_87368224735857
// MI455X (gfx1250) — matrix operand order repaired and checked by reading, not compiled, not run on hardware
//
#include <hip/hip_runtime.h>
#include <hip/hip_bf16.h>

// ---------------------------------------------------------------------------
// GNN encoder (2x GATConv + TopK pooling + MLP head) for MI455X / gfx1250.
// Big GEMMs go through v_wmma_f32_16x16x32_bf16 (wave32 WMMA, f32 accum);
// edge softmax/aggregation uses f32 global atomics (L2-resident working set).
// ---------------------------------------------------------------------------

#define BATCH   16
#define NPG     512          // nodes per graph
#define NN1     (BATCH*NPG)  // 8192
#define NEDGE   131072
#define FEAT    3
#define EMB     512
#define NH      3
#define HC      (NH*EMB)     // 1536
#define K1      410
#define K2      205
#define NP1     (BATCH*K1)   // 6560
#define NP2     (BATCH*K2)   // 3280

typedef __attribute__((ext_vector_type(16))) __bf16 v16bf;
typedef __attribute__((ext_vector_type(8)))  __bf16 v8bf;
typedef __attribute__((ext_vector_type(4)))  __bf16 v4bf;
typedef __attribute__((ext_vector_type(8)))  float  v8f;
typedef __attribute__((ext_vector_type(4)))  float  v4f;

// ---------------------------------------------------------------------------
// Generic fills / init
// ---------------------------------------------------------------------------
__global__ void fill_f32(float* p, float v, int n) {
    int i = blockIdx.x * blockDim.x + threadIdx.x;
    if (i < n) p[i] = v;
}
__global__ void fill_i32(int* p, int v, int n) {
    int i = blockIdx.x * blockDim.x + threadIdx.x;
    if (i < n) p[i] = v;
}
// out[r*cols+c] = bias[c]  (GAT accumulator starts at conv bias)
__global__ void bias_rows_kernel(float* out, const float* bias, int rows, int cols) {
    int i = blockIdx.x * blockDim.x + threadIdx.x;
    if (i < rows * cols) out[i] = bias[i % cols];
}
__global__ void add_vec_kernel(const float* a, const float* b, float* o, int n) {
    int i = blockIdx.x * blockDim.x + threadIdx.x;
    if (i < n) o[i] = a[i] + b[i];
}

// ---------------------------------------------------------------------------
// Layer-1 feature lift: xp[n, j] = sum_f x[n,f] * W1[f,j]   (K = 3, scalar)
// ---------------------------------------------------------------------------
__global__ void xp1_kernel(const float* __restrict__ x, const float* __restrict__ W,
                           float* __restrict__ xp, int n_nodes) {
    int t = blockIdx.x * blockDim.x + threadIdx.x;
    if (t >= n_nodes * HC) return;
    int node = t / HC, j = t - node * HC;
    const float* xr = x + node * FEAT;
    xp[t] = xr[0] * W[j] + xr[1] * W[HC + j] + xr[2] * W[2 * HC + j];
}

// ---------------------------------------------------------------------------
// Per-(node, head) attention scores: a_s = <xp[n,h,:], att_s[h,:]> (wave reduce)
// ---------------------------------------------------------------------------
__global__ void att_scores_kernel(const float* __restrict__ xp,
                                  const float* __restrict__ att_s,
                                  const float* __restrict__ att_d,
                                  float* __restrict__ as_, float* __restrict__ ad_,
                                  int n_nodes) {
    int widx = (blockIdx.x * blockDim.x + threadIdx.x) >> 5;
    int lane = threadIdx.x & 31;
    if (widx >= n_nodes * NH) return;
    int node = widx / NH, h = widx - node * NH;
    const float* xr = xp + (size_t)node * HC + h * EMB;
    float ss = 0.f, sd = 0.f;
    for (int c = lane; c < EMB; c += 32) {
        float v = xr[c];
        ss += v * att_s[h * EMB + c];
        sd += v * att_d[h * EMB + c];
    }
    for (int off = 16; off > 0; off >>= 1) {
        ss += __shfl_xor(ss, off, 32);
        sd += __shfl_xor(sd, off, 32);
    }
    if (lane == 0) { as_[widx] = ss; ad_[widx] = sd; }
}

// ---------------------------------------------------------------------------
// Edge helpers
// ---------------------------------------------------------------------------
__device__ __forceinline__ void edge_endpoints(int i, int E_, const int* src0, const int* tgt0,
                                               const int* mapping, int& s, int& t, bool& valid) {
    if (i < E_) {
        s = src0[i]; t = tgt0[i];
        if (mapping) { s = mapping[s]; t = mapping[t]; valid = (s >= 0) && (t >= 0); }
        else valid = true;
    } else { s = t = i - E_; valid = true; }
}

__device__ __forceinline__ void atomicMaxF(float* addr, float v) {
    if (v >= 0.f) atomicMax((int*)addr, __float_as_int(v));
    else          atomicMin((unsigned int*)addr, __float_as_uint(v));
}

// pass 1: e = leaky_relu(a_s[src]+a_d[tgt]); segment max over targets
__global__ void edge_max_kernel(const int* __restrict__ src0, const int* __restrict__ tgt0,
                                const int* __restrict__ mapping,
                                const float* __restrict__ as_, const float* __restrict__ ad_,
                                float* __restrict__ ebuf, float* __restrict__ emax,
                                int E_, int n_nodes) {
    int i = blockIdx.x * blockDim.x + threadIdx.x;
    if (i >= E_ + n_nodes) return;
    int s, t; bool valid;
    edge_endpoints(i, E_, src0, tgt0, mapping, s, t, valid);
    if (!valid) return;
    for (int h = 0; h < NH; ++h) {
        float x = as_[s * NH + h] + ad_[t * NH + h];
        float e = (x > 0.f) ? x : 0.2f * x;
        ebuf[(size_t)i * NH + h] = e;
        atomicMaxF(&emax[t * NH + h], e);
    }
}

// pass 2: ex = exp(e - emax[tgt]); segment sum -> den; store ex in place
__global__ void edge_exp_kernel(const int* __restrict__ src0, const int* __restrict__ tgt0,
                                const int* __restrict__ mapping,
                                float* __restrict__ ebuf, const float* __restrict__ emax,
                                float* __restrict__ den, int E_, int n_nodes) {
    int i = blockIdx.x * blockDim.x + threadIdx.x;
    if (i >= E_ + n_nodes) return;
    int s, t; bool valid;
    edge_endpoints(i, E_, src0, tgt0, mapping, s, t, valid);
    if (!valid) return;
    for (int h = 0; h < NH; ++h) {
        float ex = __expf(ebuf[(size_t)i * NH + h] - emax[t * NH + h]);
        ebuf[(size_t)i * NH + h] = ex;
        atomicAdd(&den[t * NH + h], ex);
    }
}

// pass 3 (wave per edge): out[tgt, :] += alpha_h * xp[src, :]
__global__ void edge_aggr_kernel(const int* __restrict__ src0, const int* __restrict__ tgt0,
                                 const int* __restrict__ mapping,
                                 const float* __restrict__ ebuf, const float* __restrict__ den,
                                 const float* __restrict__ xp, float* __restrict__ out,
                                 int E_, int n_nodes) {
    int widx = (blockIdx.x * blockDim.x + threadIdx.x) >> 5;
    int lane = threadIdx.x & 31;
    if (widx >= E_ + n_nodes) return;
    int s, t; bool valid;
    edge_endpoints(widx, E_, src0, tgt0, mapping, s, t, valid);
    if (!valid) return;
    float a0 = ebuf[(size_t)widx * NH + 0] / fmaxf(den[t * NH + 0], 1e-16f);
    float a1 = ebuf[(size_t)widx * NH + 1] / fmaxf(den[t * NH + 1], 1e-16f);
    float a2 = ebuf[(size_t)widx * NH + 2] / fmaxf(den[t * NH + 2], 1e-16f);
    const float* xs = xp + (size_t)s * HC;
    float*       ot = out + (size_t)t * HC;
    for (int c = lane; c < HC; c += 32) {
        float al = (c < EMB) ? a0 : ((c < 2 * EMB) ? a1 : a2);
        atomicAdd(&ot[c], al * xs[c]);
    }
}

// ---------------------------------------------------------------------------
// bf16 WMMA GEMM: C[M,N] = act(A[M,K] @ B[K,N] + bias)
// 128x128 block tile, 8 waves; wave w owns rows [16w,16w+16) x all 128 cols.
// Staging: A via coalesced b128 loads -> b64 LDS stores; B via coalesced b32
// column loads -> 2x b128 LDS stores in transposed layout. LDS rows padded to
// 80B (16B-aligned, 16 distinct banks across 16 lanes on fragment loads).
// ---------------------------------------------------------------------------
#define BM 128
#define BN 128
#define BK 32
#define LDT 40   // padded LDS row stride in bf16 elements (80 bytes)

__global__ __launch_bounds__(256)
void gemm_bf16_wmma(const float* __restrict__ A, const float* __restrict__ B,
                    const float* __restrict__ bias, float* __restrict__ C,
                    int M, int N, int K, int relu) {
    __shared__ __bf16 As[BM][LDT];   // 10 KB
    __shared__ __bf16 Bs[BN][LDT];   // 10 KB, K contiguous per column
    const int tid  = threadIdx.x;
    const int wave = tid >> 5;
    const int lane = tid & 31;
    const int half = lane >> 4;     // 0: lanes 0-15, 1: lanes 16-31
    const int r16  = lane & 15;
    const int m0 = blockIdx.y * BM;
    const int n0 = blockIdx.x * BN;

    // B staging coordinates (fixed per thread)
    const int bcol = tid & 127;            // column within BN
    const int bkb  = (tid >> 7) * 16;      // 16 K-rows per thread
    const bool bok = (n0 + bcol < N);

    v8f acc[8];
    const v8f vzero = {0.f, 0.f, 0.f, 0.f, 0.f, 0.f, 0.f, 0.f};
    #pragma unroll
    for (int i = 0; i < 8; ++i) acc[i] = vzero;

    for (int k0 = 0; k0 < K; k0 += BK) {
        // ---- stage A tile: 4 chunks of float4 per thread (coalesced b128) ----
        #pragma unroll
        for (int i = 0; i < 4; ++i) {
            int l   = (tid + i * 256) * 4;        // element index in 128x32 tile
            int row = l >> 5;
            int col = l & (BK - 1);
            v4f v;
            if (m0 + row < M) {
                v = *(const v4f*)(A + (size_t)(m0 + row) * K + k0 + col);
            } else {
                v = (v4f){0.f, 0.f, 0.f, 0.f};
            }
            v4bf b;
            b[0] = (__bf16)v[0]; b[1] = (__bf16)v[1];
            b[2] = (__bf16)v[2]; b[3] = (__bf16)v[3];
            *(v4bf*)&As[row][col] = b;            // ds_store_b64
        }
        // ---- stage B tile transposed: 16 coalesced b32 column loads/thread ----
        {
            v8bf lo, hi;
            #pragma unroll
            for (int j = 0; j < 8; ++j) {
                float v = bok ? B[(size_t)(k0 + bkb + j) * N + n0 + bcol] : 0.f;
                lo[j] = (__bf16)v;
            }
            #pragma unroll
            for (int j = 0; j < 8; ++j) {
                float v = bok ? B[(size_t)(k0 + bkb + 8 + j) * N + n0 + bcol] : 0.f;
                hi[j] = (__bf16)v;
            }
            *(v8bf*)&Bs[bcol][bkb]     = lo;      // ds_store_b128
            *(v8bf*)&Bs[bcol][bkb + 8] = hi;      // ds_store_b128
        }
        __syncthreads();

        // A fragment (16x32): lanes 0-15 hold K {0..7,16..23}, lanes 16-31 {8..15,24..31}
        const __bf16* ap = &As[wave * 16 + r16][0];
        v8bf a_lo = *(const v8bf*)(ap + half * 8);
        v8bf a_hi = *(const v8bf*)(ap + 16 + half * 8);
        v16bf afrag = __builtin_shufflevector(a_lo, a_hi,
            0, 1, 2, 3, 4, 5, 6, 7, 8, 9, 10, 11, 12, 13, 14, 15);

        #pragma unroll
        for (int nt = 0; nt < 8; ++nt) {
            // B fragment (32x16): lanes 0-15 hold K {0..7,16..23}, lanes 16-31 {8..15,24..31}
            const __bf16* bp = &Bs[nt * 16 + r16][0];
            v8bf b_lo = *(const v8bf*)(bp + half * 8);
            v8bf b_hi = *(const v8bf*)(bp + 16 + half * 8);
            v16bf bfrag = __builtin_shufflevector(b_lo, b_hi,
                0, 1, 2, 3, 4, 5, 6, 7, 8, 9, 10, 11, 12, 13, 14, 15);
            acc[nt] = __builtin_amdgcn_wmma_f32_16x16x32_bf16(
                false, afrag, false, bfrag, (short)0, acc[nt], false, false);
        }
        __syncthreads();
    }

    // store: VGPR r, lane l -> M = r + 8*(l>>4), N = l&15 within 16x16 tile
    #pragma unroll
    for (int nt = 0; nt < 8; ++nt) {
        int col = n0 + nt * 16 + r16;
        if (col >= N) continue;
        float bv = bias ? bias[col] : 0.f;
        #pragma unroll
        for (int rr = 0; rr < 8; ++rr) {
            int row = m0 + wave * 16 + half * 8 + rr;
            if (row < M) {
                float v = acc[nt][rr] + bv;
                if (relu) v = fmaxf(v, 0.f);
                C[(size_t)row * N + col] = v;
            }
        }
    }
}

// ---------------------------------------------------------------------------
// BatchNorm (training stats, biased var), block per column, in place
// ---------------------------------------------------------------------------
__global__ void bn_kernel(float* __restrict__ h, const float* __restrict__ g,
                          const float* __restrict__ b, int rows) {
    int c = blockIdx.x;  // 0..EMB-1
    __shared__ float s1[256], s2[256];
    __shared__ float scale_sh, shift_sh;
    float sum = 0.f, sq = 0.f;
    for (int r = threadIdx.x; r < rows; r += 256) {
        float v = h[(size_t)r * EMB + c];
        sum += v; sq += v * v;
    }
    s1[threadIdx.x] = sum; s2[threadIdx.x] = sq; __syncthreads();
    for (int off = 128; off > 0; off >>= 1) {
        if (threadIdx.x < off) {
            s1[threadIdx.x] += s1[threadIdx.x + off];
            s2[threadIdx.x] += s2[threadIdx.x + off];
        }
        __syncthreads();
    }
    if (threadIdx.x == 0) {
        float m = s1[0] / rows;
        float v = s2[0] / rows - m * m;
        float inv = rsqrtf(v + 1e-5f);
        scale_sh = g[c] * inv;
        shift_sh = b[c] - m * g[c] * inv;
    }
    __syncthreads();
    float sc = scale_sh, sh = shift_sh;
    for (int r = threadIdx.x; r < rows; r += 256)
        h[(size_t)r * EMB + c] = h[(size_t)r * EMB + c] * sc + sh;
}

// ---------------------------------------------------------------------------
// TopK pooling pieces
// ---------------------------------------------------------------------------
__global__ void pnorm_kernel(const float* __restrict__ p, int n, float* __restrict__ invn) {
    __shared__ float s[256];
    float acc = 0.f;
    for (int i = threadIdx.x; i < n; i += 256) acc += p[i] * p[i];
    s[threadIdx.x] = acc; __syncthreads();
    for (int off = 128; off > 0; off >>= 1) {
        if (threadIdx.x < off) s[threadIdx.x] += s[threadIdx.x + off];
        __syncthreads();
    }
    if (threadIdx.x == 0) invn[0] = rsqrtf(s[0]);
}

__global__ void score_kernel(const float* __restrict__ h, const float* __restrict__ p,
                             const float* __restrict__ invn, float* __restrict__ score,
                             int n_nodes) {
    int idx  = (blockIdx.x * blockDim.x + threadIdx.x) >> 5;
    int lane = threadIdx.x & 31;
    if (idx >= n_nodes) return;
    float s = 0.f;
    for (int c = lane; c < EMB; c += 32) s += h[(size_t)idx * EMB + c] * p[c];
    for (int off = 16; off > 0; off >>= 1) s += __shfl_xor(s, off, 32);
    if (lane == 0) score[idx] = tanhf(s * invn[0]);
}

// block per graph: bitonic sort (desc) of up to 512 (score,idx) pairs in LDS
__global__ void topk_kernel(const float* __restrict__ score, float* __restrict__ tops,
                            int* __restrict__ perm, int* __restrict__ mapping,
                            int n_per, int k) {
    __shared__ float sk[512];
    __shared__ int   si[512];
    int g = blockIdx.x, tid = threadIdx.x;
    for (int j = tid; j < 512; j += 256) {
        sk[j] = (j < n_per) ? score[g * n_per + j] : -3.4e38f;
        si[j] = j;
    }
    __syncthreads();
    for (int ksz = 2; ksz <= 512; ksz <<= 1) {
        for (int jj = ksz >> 1; jj > 0; jj >>= 1) {
            for (int i = tid; i < 512; i += 256) {
                int ixj = i ^ jj;
                if (ixj > i) {
                    bool desc = ((i & ksz) == 0);
                    float a = sk[i], b = sk[ixj];
                    if (desc ? (a < b) : (a > b)) {
                        sk[i] = b; sk[ixj] = a;
                        int t = si[i]; si[i] = si[ixj]; si[ixj] = t;
                    }
                }
            }
            __syncthreads();
        }
    }
    for (int j = tid; j < k; j += 256) {
        int oldid = g * n_per + si[j];
        int newid = g * k + j;
        tops[newid] = sk[j];
        perm[newid] = oldid;
        if (mapping) mapping[oldid] = newid;
    }
}

// block per pooled row: out[row,:] = h[perm[row],:] * tops[row]
__global__ void pool_gather_kernel(const float* __restrict__ h, const int* __restrict__ perm,
                                   const float* __restrict__ tops, float* __restrict__ out,
                                   int nrows) {
    int row = blockIdx.x;
    if (row >= nrows) return;
    int old = perm[row];
    float s = tops[row];
    for (int c = threadIdx.x; c < EMB; c += blockDim.x)
        out[(size_t)row * EMB + c] = h[(size_t)old * EMB + c] * s;
}

// [global_max_pool, global_mean_pool] -> xout[g, 0:512 | 512:1024]
__global__ void pool_cat_kernel(const float* __restrict__ hp, float* __restrict__ xout, int k) {
    int t = blockIdx.x * blockDim.x + threadIdx.x;
    if (t >= BATCH * EMB) return;
    int g = t >> 9, c = t & (EMB - 1);
    float mx = -3.4e38f, sm = 0.f;
    for (int j = 0; j < k; ++j) {
        float v = hp[(size_t)(g * k + j) * EMB + c];
        mx = fmaxf(mx, v); sm += v;
    }
    xout[g * 2 * EMB + c]       = mx;
    xout[g * 2 * EMB + EMB + c] = sm / (float)k;
}

// ---------------------------------------------------------------------------
// Host launch
// ---------------------------------------------------------------------------
static inline dim3 cdiv256(long long n) { return dim3((unsigned)((n + 255) / 256)); }

extern "C" void kernel_launch(void* const* d_in, const int* in_sizes, int n_in,
                              void* d_out, int out_size, void* d_ws, size_t ws_size,
                              hipStream_t stream) {
    const float* x    = (const float*)d_in[0];
    const int*   ei   = (const int*)  d_in[1];
    const float* W1   = (const float*)d_in[2];
    const float* as1  = (const float*)d_in[3];
    const float* ad1  = (const float*)d_in[4];
    const float* bc1  = (const float*)d_in[5];
    const float* Wh1  = (const float*)d_in[6];
    const float* bh1  = (const float*)d_in[7];
    const float* g1   = (const float*)d_in[8];
    const float* be1  = (const float*)d_in[9];
    const float* p1   = (const float*)d_in[10];
    const float* W2   = (const float*)d_in[11];
    const float* as2  = (const float*)d_in[12];
    const float* ad2  = (const float*)d_in[13];
    const float* bc2  = (const float*)d_in[14];
    const float* Wh2  = (const float*)d_in[15];
    const float* bh2  = (const float*)d_in[16];
    const float* g2   = (const float*)d_in[17];
    const float* be2  = (const float*)d_in[18];
    const float* p2   = (const float*)d_in[19];
    const float* Wl1  = (const float*)d_in[20];
    const float* bl1  = (const float*)d_in[21];
    const float* Wl2  = (const float*)d_in[22];
    const float* bl2  = (const float*)d_in[23];
    const int* src0 = ei;
    const int* tgt0 = ei + NEDGE;

    // scratch carve-out (~140 MB)
    size_t off = 0;
    auto alloc = [&](size_t bytes) -> void* {
        void* p = (char*)d_ws + off;
        off += (bytes + 255) & ~(size_t)255;
        return p;
    };
    float* xp      = (float*)alloc((size_t)NN1 * HC * 4);
    float* gat     = (float*)alloc((size_t)NN1 * HC * 4);
    float* hbuf    = (float*)alloc((size_t)NN1 * EMB * 4);
    float* hp1     = (float*)alloc((size_t)NP1 * EMB * 4);
    float* hp2     = (float*)alloc((size_t)NP2 * EMB * 4);
    float* asb     = (float*)alloc((size_t)NN1 * NH * 4);
    float* adb     = (float*)alloc((size_t)NN1 * NH * 4);
    float* emax    = (float*)alloc((size_t)NN1 * NH * 4);
    float* den     = (float*)alloc((size_t)NN1 * NH * 4);
    float* ebuf    = (float*)alloc((size_t)(NEDGE + NN1) * NH * 4);
    float* scoreb  = (float*)alloc((size_t)NN1 * 4);
    float* tops    = (float*)alloc((size_t)NP1 * 4);
    int*   perm    = (int*)  alloc((size_t)NP1 * 4);
    int*   mapping = (int*)  alloc((size_t)NN1 * 4);
    float* x1      = (float*)alloc((size_t)BATCH * 2 * EMB * 4);
    float* x2      = (float*)alloc((size_t)BATCH * 2 * EMB * 4);
    float* xg      = (float*)alloc((size_t)BATCH * 2 * EMB * 4);
    float* t1      = (float*)alloc((size_t)BATCH * EMB * 4);
    float* invn    = (float*)alloc(256);

    const int NE1 = NEDGE + NN1;
    const int NE2 = NEDGE + NP1;

    // ---------------- Layer 1 ----------------
    xp1_kernel<<<cdiv256((long long)NN1 * HC), 256, 0, stream>>>(x, W1, xp, NN1);
    att_scores_kernel<<<dim3((NN1 * NH + 7) / 8), 256, 0, stream>>>(xp, as1, ad1, asb, adb, NN1);
    fill_f32<<<cdiv256(NN1 * NH), 256, 0, stream>>>(emax, -3.4e38f, NN1 * NH);
    fill_f32<<<cdiv256(NN1 * NH), 256, 0, stream>>>(den, 0.f, NN1 * NH);
    edge_max_kernel<<<cdiv256(NE1), 256, 0, stream>>>(src0, tgt0, nullptr, asb, adb, ebuf, emax, NEDGE, NN1);
    edge_exp_kernel<<<cdiv256(NE1), 256, 0, stream>>>(src0, tgt0, nullptr, ebuf, emax, den, NEDGE, NN1);
    bias_rows_kernel<<<cdiv256((long long)NN1 * HC), 256, 0, stream>>>(gat, bc1, NN1, HC);
    edge_aggr_kernel<<<dim3((NE1 + 7) / 8), 256, 0, stream>>>(src0, tgt0, nullptr, ebuf, den, xp, gat, NEDGE, NN1);
    gemm_bf16_wmma<<<dim3(EMB / BN, (NN1 + BM - 1) / BM), 256, 0, stream>>>(
        gat, Wh1, bh1, hbuf, NN1, EMB, HC, 1);
    bn_kernel<<<dim3(EMB), 256, 0, stream>>>(hbuf, g1, be1, NN1);
    pnorm_kernel<<<1, 256, 0, stream>>>(p1, EMB, invn);
    score_kernel<<<dim3((NN1 + 7) / 8), 256, 0, stream>>>(hbuf, p1, invn, scoreb, NN1);
    fill_i32<<<cdiv256(NN1), 256, 0, stream>>>(mapping, -1, NN1);
    topk_kernel<<<dim3(BATCH), 256, 0, stream>>>(scoreb, tops, perm, mapping, NPG, K1);
    pool_gather_kernel<<<dim3(NP1), 256, 0, stream>>>(hbuf, perm, tops, hp1, NP1);
    pool_cat_kernel<<<cdiv256(BATCH * EMB), 256, 0, stream>>>(hp1, x1, K1);

    // ---------------- Layer 2 ----------------
    gemm_bf16_wmma<<<dim3(HC / BN, (NP1 + BM - 1) / BM), 256, 0, stream>>>(
        hp1, W2, nullptr, xp, NP1, HC, EMB, 0);
    att_scores_kernel<<<dim3((NP1 * NH + 7) / 8), 256, 0, stream>>>(xp, as2, ad2, asb, adb, NP1);
    fill_f32<<<cdiv256(NP1 * NH), 256, 0, stream>>>(emax, -3.4e38f, NP1 * NH);
    fill_f32<<<cdiv256(NP1 * NH), 256, 0, stream>>>(den, 0.f, NP1 * NH);
    edge_max_kernel<<<cdiv256(NE2), 256, 0, stream>>>(src0, tgt0, mapping, asb, adb, ebuf, emax, NEDGE, NP1);
    edge_exp_kernel<<<cdiv256(NE2), 256, 0, stream>>>(src0, tgt0, mapping, ebuf, emax, den, NEDGE, NP1);
    bias_rows_kernel<<<cdiv256((long long)NP1 * HC), 256, 0, stream>>>(gat, bc2, NP1, HC);
    edge_aggr_kernel<<<dim3((NE2 + 7) / 8), 256, 0, stream>>>(src0, tgt0, mapping, ebuf, den, xp, gat, NEDGE, NP1);
    gemm_bf16_wmma<<<dim3(EMB / BN, (NP1 + BM - 1) / BM), 256, 0, stream>>>(
        gat, Wh2, bh2, hbuf, NP1, EMB, HC, 1);
    bn_kernel<<<dim3(EMB), 256, 0, stream>>>(hbuf, g2, be2, NP1);
    pnorm_kernel<<<1, 256, 0, stream>>>(p2, EMB, invn);
    score_kernel<<<dim3((NP1 + 7) / 8), 256, 0, stream>>>(hbuf, p2, invn, scoreb, NP1);
    topk_kernel<<<dim3(BATCH), 256, 0, stream>>>(scoreb, tops, perm, nullptr, K1, K2);
    pool_gather_kernel<<<dim3(NP2), 256, 0, stream>>>(hbuf, perm, tops, hp2, NP2);
    pool_cat_kernel<<<cdiv256(BATCH * EMB), 256, 0, stream>>>(hp2, x2, K2);

    // ---------------- Head ----------------
    add_vec_kernel<<<cdiv256(BATCH * 2 * EMB), 256, 0, stream>>>(x1, x2, xg, BATCH * 2 * EMB);
    gemm_bf16_wmma<<<dim3(EMB / BN, 1), 256, 0, stream>>>(
        xg, Wl1, bl1, t1, BATCH, EMB, 2 * EMB, 1);
    gemm_bf16_wmma<<<dim3(256 / BN, 1), 256, 0, stream>>>(
        t1, Wl2, bl2, (float*)d_out, BATCH, 256, EMB, 0);
}
